// AdaptiveKernelConv_17514876633853
// MI455X (gfx1250) — matrix operand order repaired and checked by reading, not compiled, not run on hardware
//
#include <hip/hip_runtime.h>
#include <math.h>

// ---------------- problem constants ----------------
#define Bn   16
#define Cn   128
#define On   256
#define Hn   96
#define Wn   96
#define HWn  9216      // Hn*Wn
#define TAPS 49        // 7*7
#define OFFC 98        // 2*TAPS
#define KDIM 1152      // Cn*3*3 (im2col K for offset conv)
#define MPAD 112       // OFFC padded to 7*16
#define HIDN 32
#define Sn   4
#define Gn   32
#define GN_N 73728.0f  // (On/Gn)*HWn

#define LDB  40        // padded N-stride (shorts) for transposed B tiles: 16B aligned, bank-conflict free
#define LDA_OFF 40     // padded K-stride for offset-conv A tile
#define LDA_PW  136    // padded K-stride for pw-gemm A panel (128 + 8)

typedef __attribute__((ext_vector_type(16))) __bf16          v16bf;
typedef __attribute__((ext_vector_type(8)))  float           v8f;

struct U32x8 { uint4 a, b; };   // 32 bytes, bit_cast carrier for a bf16 fragment

// float -> bf16 bits, round-to-nearest-even
__device__ __forceinline__ unsigned short f2bf(float f) {
    union { float f; unsigned u; } un; un.f = f;
    unsigned r = un.u + 0x7FFFu + ((un.u >> 16) & 1u);
    return (unsigned short)(r >> 16);
}

// CDNA5 16-bit A fragment (16x32, MxK), lane L holds row M=L%16.
// Per-lane data = two contiguous 8-short runs at kBase+kh and kBase+16+kh.
__device__ __forceinline__ v16bf load_frag_a(const unsigned short* lA, int ld, int mBase, int kBase, int lane) {
    int m  = mBase + (lane & 15);
    int kh = kBase + ((lane >> 4) << 3);
    const unsigned short* p = lA + m * ld + kh;
    U32x8 s;
    s.a = *(const uint4*)(p);
    s.b = *(const uint4*)(p + 16);
    return __builtin_bit_cast(v16bf, s);
}

// CDNA5 16-bit B fragment (32x16, KxN) from an N-major (transposed) LDS tile.
// Lane L holds col N=L%16; its 16 K-elements are two contiguous 8-short runs at kh and 16+kh.
__device__ __forceinline__ v16bf load_frag_bT(const unsigned short* lBt, int nBase, int lane) {
    int n  = nBase + (lane & 15);
    int kh = (lane >> 4) << 3;
    const unsigned short* p = lBt + n * LDB + kh;
    U32x8 s;
    s.a = *(const uint4*)(p);
    s.b = *(const uint4*)(p + 16);
    return __builtin_bit_cast(v16bf, s);
}

// ---------------- 1: pad/convert off_w to bf16 A matrix [MPAD][KDIM] ----------------
__global__ __launch_bounds__(256) void k_prep_offw(const float* __restrict__ off_w,
                                                   unsigned short* __restrict__ A_off) {
    int idx = blockIdx.x * 256 + threadIdx.x;
    if (idx >= MPAD * KDIM) return;
    int m = idx / KDIM, k = idx - m * KDIM;
    float v = (m < OFFC) ? off_w[m * KDIM + k] : 0.0f;
    A_off[idx] = f2bf(v);
}

// ---------------- 2: per-(b,c) spatial mean ----------------
__global__ __launch_bounds__(256) void k_mean(const float* __restrict__ x, float* __restrict__ gmean) {
    __shared__ float red[256];
    int tid = threadIdx.x;
    int bc  = blockIdx.x;
    const float* xp = x + (size_t)bc * HWn;
    float s = 0.0f;
    for (int i = tid; i < HWn; i += 256) s += xp[i];
    red[tid] = s; __syncthreads();
    for (int off = 128; off > 0; off >>= 1) {
        if (tid < off) red[tid] += red[tid + off];
        __syncthreads();
    }
    if (tid == 0) gmean[bc] = red[0] * (1.0f / (float)HWn);
}

// ---------------- 3: gating MLP + softmax + w_eff (bf16) ----------------
__global__ __launch_bounds__(256) void k_gates_weff(const float* __restrict__ gmean,
                                                    const float* __restrict__ gate_w1,
                                                    const float* __restrict__ gate_w2,
                                                    const float* __restrict__ proj_w,
                                                    const float* __restrict__ pw_w,
                                                    unsigned short* __restrict__ w_eff) {
    __shared__ float gl[Cn];
    __shared__ float hid[HIDN];
    __shared__ float gates[Sn];
    int tid = threadIdx.x;
    int b   = blockIdx.x;
    if (tid < Cn) gl[tid] = gmean[b * Cn + tid];
    __syncthreads();
    if (tid < HIDN) {
        float s = 0.0f;
        for (int c = 0; c < Cn; ++c) s += gl[c] * gate_w1[tid * Cn + c];
        hid[tid] = fmaxf(s, 0.0f);
    }
    __syncthreads();
    if (tid < Sn) {
        float s = 0.0f;
        for (int i = 0; i < HIDN; ++i) s += hid[i] * gate_w2[tid * HIDN + i];
        gates[tid] = s;
    }
    __syncthreads();
    if (tid == 0) {
        float m = fmaxf(fmaxf(gates[0], gates[1]), fmaxf(gates[2], gates[3]));
        float e0 = expf(gates[0] - m), e1 = expf(gates[1] - m);
        float e2 = expf(gates[2] - m), e3 = expf(gates[3] - m);
        float inv = 1.0f / (e0 + e1 + e2 + e3);
        gates[0] = e0 * inv; gates[1] = e1 * inv; gates[2] = e2 * inv; gates[3] = e3 * inv;
    }
    __syncthreads();
    for (int idx = tid; idx < On * Cn; idx += 256) {
        float v = pw_w[idx];
#pragma unroll
        for (int s = 0; s < Sn; ++s) v += gates[s] * proj_w[s * (On * Cn) + idx];
        w_eff[(size_t)b * On * Cn + idx] = f2bf(v);
    }
}

// ---------------- 4: offset conv as WMMA implicit GEMM ----------------
// per block: b, 64-pixel tile; M=112 (7 waves x 16), K=1152 in 36 chunks of 32.
// Output padded to MPAD rows -> branchless epilogue.
__global__ __launch_bounds__(256) void k_offset_conv(const float* __restrict__ x,
                                                     const unsigned short* __restrict__ A_off,
                                                     const float* __restrict__ off_b,
                                                     float* __restrict__ offbuf) {
    __shared__ __align__(16) unsigned short lA[MPAD * LDA_OFF];
    __shared__ __align__(16) unsigned short lBt[64 * LDB];     // transposed: [n][k]
    __shared__ float lOffB[MPAD];
    int tid  = threadIdx.x;
    int lane = tid & 31;
    int wave = tid >> 5;
    int tile = blockIdx.x;
    int b    = blockIdx.y;
    int pixBase = tile * 64;

    if (tid < MPAD) lOffB[tid] = (tid < OFFC) ? off_b[tid] : 0.0f;

    v8f acc[4];
#pragma unroll
    for (int n = 0; n < 4; ++n)
#pragma unroll
        for (int e = 0; e < 8; ++e) acc[n][e] = 0.0f;

    for (int ks = 0; ks < 36; ++ks) {
        int kBase = ks * 32;
        // A tile: 112x32 shorts = 448 uint4, vectorized
        for (int i = tid; i < (MPAD * 32) / 8; i += 256) {
            int m = i >> 2, q = i & 3;
            const uint4* src = (const uint4*)(A_off + m * KDIM + kBase);
            ((uint4*)(lA + m * LDA_OFF))[q] = src[q];
        }
        // B tile (im2col, stored transposed): 32x64 elements
        for (int i = tid; i < 32 * 64; i += 256) {
            int kk = i >> 6, j = i & 63;
            int k = kBase + kk;
            int c  = k / 9;  int r9 = k - 9 * c;
            int ky = r9 / 3; int kx = r9 - 3 * ky;
            int pix = pixBase + j;
            int h = pix / 96, w = pix - 96 * h;
            int y = h + ky - 1, xw = w + kx - 1;
            float v = 0.0f;
            if (y >= 0 && y < Hn && xw >= 0 && xw < Wn)
                v = x[(((size_t)b * Cn + c) * Hn + y) * Wn + xw];
            lBt[j * LDB + kk] = f2bf(v);
        }
        __syncthreads();
        if (wave < 7) {
            v16bf af = load_frag_a(lA, LDA_OFF, wave * 16, 0, lane);
#pragma unroll
            for (int n = 0; n < 4; ++n) {
                v16bf bf = load_frag_bT(lBt, n * 16, lane);
                acc[n] = __builtin_amdgcn_wmma_f32_16x16x32_bf16(
                    false, af, false, bf, (short)0, acc[n], false, false);
            }
        }
        __syncthreads();
    }

    if (wave < 7) {
        int mhalf = (lane >> 4) << 3;         // 0 or 8
        int m0 = wave * 16 + mhalf;
#pragma unroll
        for (int n = 0; n < 4; ++n) {
            float* op = offbuf + ((size_t)b * MPAD + m0) * HWn + pixBase + n * 16 + (lane & 15);
#pragma unroll
            for (int r = 0; r < 8; ++r)
                op[(size_t)r * HWn] = acc[n][r] + lOffB[m0 + r];
        }
    }
}

// ---------------- 5: deformable depthwise conv -> bf16 ----------------
__global__ __launch_bounds__(256) void k_deform(const float* __restrict__ x,
                                                const float* __restrict__ dw_weight,
                                                const float* __restrict__ offbuf,
                                                unsigned short* __restrict__ dwb) {
    __shared__ float lW[TAPS * Cn];       // [tap][channel]
    int tid = threadIdx.x;
    int b   = blockIdx.y;
    int pix = blockIdx.x * 256 + tid;
    for (int i = tid; i < TAPS * Cn; i += 256) {
        int t = i >> 7, c = i & 127;
        lW[i] = dw_weight[c * TAPS + t];  // dw_weight (C,1,7,7)
    }
    __syncthreads();
    int h = pix / 96, w = pix - 96 * h;
    const float* xb = x + (size_t)b * Cn * HWn;
    const float* ob = offbuf + (size_t)b * MPAD * HWn;

    for (int cb = 0; cb < Cn; cb += 16) {
        float acc[16];
#pragma unroll
        for (int i = 0; i < 16; ++i) acc[i] = 0.0f;
        for (int t = 0; t < TAPS; ++t) {
            float oy = ob[(size_t)(2 * t) * HWn + pix];
            float ox = ob[(size_t)(2 * t + 1) * HWn + pix];
            float py = (float)(h - 3 + t / 7) + oy;
            float px = (float)(w - 3 + t % 7) + ox;
            float y0f = floorf(py), x0f = floorf(px);
            float wy = py - y0f, wx = px - x0f;
            int y0 = (int)y0f, x0 = (int)x0f;
            int y1 = y0 + 1,  x1 = x0 + 1;
            float vy0 = (y0 >= 0 && y0 < Hn) ? 1.0f : 0.0f;
            float vy1 = (y1 >= 0 && y1 < Hn) ? 1.0f : 0.0f;
            float vx0 = (x0 >= 0 && x0 < Wn) ? 1.0f : 0.0f;
            float vx1 = (x1 >= 0 && x1 < Wn) ? 1.0f : 0.0f;
            int y0c = min(max(y0, 0), Hn - 1), y1c = min(max(y1, 0), Hn - 1);
            int x0c = min(max(x0, 0), Wn - 1), x1c = min(max(x1, 0), Wn - 1);
            float w00 = (1.0f - wy) * (1.0f - wx) * vy0 * vx0;
            float w01 = (1.0f - wy) * wx * vy0 * vx1;
            float w10 = wy * (1.0f - wx) * vy1 * vx0;
            float w11 = wy * wx * vy1 * vx1;
            int i00 = y0c * Wn + x0c, i01 = y0c * Wn + x1c;
            int i10 = y1c * Wn + x0c, i11 = y1c * Wn + x1c;
            const float* wt = &lW[t * Cn + cb];
#pragma unroll
            for (int cc = 0; cc < 16; ++cc) {
                const float* xp = xb + (size_t)(cb + cc) * HWn;
                acc[cc] += wt[cc] * (w00 * xp[i00] + w01 * xp[i01] + w10 * xp[i10] + w11 * xp[i11]);
            }
        }
#pragma unroll
        for (int cc = 0; cc < 16; ++cc)
            dwb[((size_t)b * Cn + cb + cc) * HWn + pix] = f2bf(acc[cc]);
    }
}

// ---------------- 6: adaptive pointwise WMMA GEMM + fused GN stats ----------------
// per block: b, half (o-range of 128), 64-pixel tile. A panel staged once; K loop restages B only.
__global__ __launch_bounds__(256) void k_pw_gemm(const unsigned short* __restrict__ w_eff,
                                                 const unsigned short* __restrict__ dwb,
                                                 float* __restrict__ out,
                                                 float* __restrict__ gsum,
                                                 float* __restrict__ gsq) {
    __shared__ __align__(16) unsigned short lA[128 * LDA_PW];  // full 128x128 A panel
    __shared__ __align__(16) unsigned short lBt[64 * LDB];     // transposed B tile [n][k]
    __shared__ float gsL[16], gqL[16];
    int tid  = threadIdx.x;
    int lane = tid & 31;
    int wave = tid >> 5;
    int tile = blockIdx.x;
    int half = blockIdx.y;
    int b    = blockIdx.z;
    int pixBase = tile * 64;
    if (tid < 16) { gsL[tid] = 0.0f; gqL[tid] = 0.0f; }

    v8f acc[4];
#pragma unroll
    for (int n = 0; n < 4; ++n)
#pragma unroll
        for (int e = 0; e < 8; ++e) acc[n][e] = 0.0f;

    const unsigned short* Ab = w_eff + ((size_t)b * On + half * 128) * Cn;
    const unsigned short* Bb = dwb + (size_t)b * Cn * HWn;

    // stage full A panel once: 128x128 shorts = 2048 uint4
    {
        const uint4* Ab4 = (const uint4*)Ab;
        for (int i = tid; i < 2048; i += 256) {
            int m = i >> 4, q = i & 15;
            ((uint4*)(lA + m * LDA_PW))[q] = Ab4[m * 16 + q];
        }
    }

    for (int ks = 0; ks < 4; ++ks) {
        int kBase = ks * 32;
        // B tile: vector-load 8 shorts/thread, scatter transposed into LDS
        {
            int kk = tid >> 3, jv = (tid & 7) * 8;
            union { uint4 v; unsigned short u[8]; } ld;
            ld.v = *(const uint4*)(Bb + (size_t)(kBase + kk) * HWn + pixBase + jv);
#pragma unroll
            for (int e = 0; e < 8; ++e)
                lBt[(jv + e) * LDB + kk] = ld.u[e];
        }
        __syncthreads();
        v16bf af = load_frag_a(lA, LDA_PW, wave * 16, kBase, lane);
#pragma unroll
        for (int n = 0; n < 4; ++n) {
            v16bf bf = load_frag_bT(lBt, n * 16, lane);
            acc[n] = __builtin_amdgcn_wmma_f32_16x16x32_bf16(
                false, af, false, bf, (short)0, acc[n], false, false);
        }
        __syncthreads();
    }

    // epilogue: branchless stores with immediate offsets + group-norm partials
    int mhalf = (lane >> 4) << 3;  // 0 or 8
    int o0 = half * 128 + wave * 16 + mhalf;
    float s = 0.0f, q = 0.0f;
#pragma unroll
    for (int n = 0; n < 4; ++n) {
        float* op = out + ((size_t)b * On + o0) * HWn + pixBase + n * 16 + (lane & 15);
#pragma unroll
        for (int r = 0; r < 8; ++r) {
            float v = acc[n][r];
            op[(size_t)r * HWn] = v;
            s += v; q += v * v;
        }
    }
    // all 32 values of a lane land in one local group: wave*2 + (lane>=16)
    int gl = wave * 2 + (lane >> 4);
    atomicAdd(&gsL[gl], s);
    atomicAdd(&gqL[gl], q);
    __syncthreads();
    if (tid < 16) {
        atomicAdd(&gsum[b * Gn + half * 16 + tid], gsL[tid]);
        atomicAdd(&gsq[b * Gn + half * 16 + tid],  gqL[tid]);
    }
}

// ---------------- 7: GroupNorm + exact GELU (in place on d_out) ----------------
__global__ __launch_bounds__(256) void k_gn_gelu(float* __restrict__ out,
                                                 const float* __restrict__ gsum,
                                                 const float* __restrict__ gsq,
                                                 const float* __restrict__ gn_w,
                                                 const float* __restrict__ gn_b) {
    size_t idx = (size_t)blockIdx.x * 256 + threadIdx.x;
    int b   = (int)(idx / ((size_t)On * HWn));
    int rem = (int)(idx - (size_t)b * On * HWn);
    int o   = rem / HWn;
    int g   = o >> 3;
    float invN = 1.0f / GN_N;
    float mu  = gsum[b * Gn + g] * invN;
    float var = gsq[b * Gn + g] * invN - mu * mu;
    float v = out[idx];
    float y = (v - mu) * rsqrtf(fmaxf(var, 0.0f) + 1e-5f) * gn_w[o] + gn_b[o];
    out[idx] = 0.5f * y * (1.0f + erff(y * 0.70710678118654752f));
}

// ---------------- launch ----------------
extern "C" void kernel_launch(void* const* d_in, const int* in_sizes, int n_in,
                              void* d_out, int out_size, void* d_ws, size_t ws_size,
                              hipStream_t stream) {
    (void)in_sizes; (void)n_in; (void)out_size; (void)ws_size;
    const float* x         = (const float*)d_in[0];
    const float* dw_weight = (const float*)d_in[1];
    const float* pw_w      = (const float*)d_in[2];
    const float* off_w     = (const float*)d_in[3];
    const float* off_b     = (const float*)d_in[4];
    const float* gate_w1   = (const float*)d_in[5];
    const float* gate_w2   = (const float*)d_in[6];
    const float* proj_w    = (const float*)d_in[7];
    const float* gn_w      = (const float*)d_in[8];
    const float* gn_b      = (const float*)d_in[9];
    float* out = (float*)d_out;

    char* p = (char*)d_ws;
    auto alloc = [&](size_t bytes) {
        char* r = p;
        p += (bytes + 255) & ~(size_t)255;
        return r;
    };
    unsigned short* A_off  = (unsigned short*)alloc((size_t)MPAD * KDIM * 2);
    float*          gmean  = (float*)alloc((size_t)Bn * Cn * 4);
    unsigned short* w_eff  = (unsigned short*)alloc((size_t)Bn * On * Cn * 2);
    float*          offbuf = (float*)alloc((size_t)Bn * MPAD * HWn * 4);   // padded to MPAD rows
    unsigned short* dwb    = (unsigned short*)alloc((size_t)Bn * Cn * HWn * 2);
    float*          gstats = (float*)alloc((size_t)2 * Bn * Gn * 4);
    float* gsum = gstats;
    float* gsq  = gstats + Bn * Gn;

    k_prep_offw<<<(MPAD * KDIM + 255) / 256, 256, 0, stream>>>(off_w, A_off);
    k_mean<<<Bn * Cn, 256, 0, stream>>>(x, gmean);
    k_gates_weff<<<Bn, 256, 0, stream>>>(gmean, gate_w1, gate_w2, proj_w, pw_w, w_eff);
    k_offset_conv<<<dim3(HWn / 64, Bn), 256, 0, stream>>>(x, A_off, off_b, offbuf);
    k_deform<<<dim3(HWn / 256, Bn), 256, 0, stream>>>(x, dw_weight, offbuf, dwb);
    hipMemsetAsync(gstats, 0, (size_t)2 * Bn * Gn * 4, stream);
    k_pw_gemm<<<dim3(HWn / 64, 2, Bn), 256, 0, stream>>>(w_eff, dwb, out, gsum, gsq);
    k_gn_gelu<<<(size_t)Bn * On * HWn / 256, 256, 0, stream>>>(out, gsum, gsq, gn_w, gn_b);
}
